// ChristophCNNwithPCEN_24146306138413
// MI455X (gfx1250) — matrix operand order repaired and checked by reading, compiled, not run on hardware
//
#include <hip/hip_runtime.h>
#include <hip/hip_bf16.h>
#include <math.h>

typedef __attribute__((ext_vector_type(16))) _Float16 v16h;
typedef __attribute__((ext_vector_type(8)))  float    v8f;
typedef __attribute__((ext_vector_type(4)))  int      v4i;

// Padded per-image strides (halves), 16B-aligned for B128 paths.
#define P1STRIDE 12616ull   // pool1: 15*29*29 = 12615 -> pad 1
#define P2STRIDE 3632ull    // pool2/flatten: 30*11*11 = 3630 -> pad 2

// CDNA5 async global->LDS copy engine (ASYNCcnt), if the toolchain exposes it.
#if __has_builtin(__builtin_amdgcn_global_load_async_to_lds_b128) && \
    __has_builtin(__builtin_amdgcn_s_wait_asynccnt)
#define HAVE_ASYNC_LDS 1
#else
#define HAVE_ASYNC_LDS 0
#endif

#if HAVE_ASYNC_LDS
typedef v4i __attribute__((address_space(1))) gv4i;   // global int4
typedef v4i __attribute__((address_space(3))) lv4i;   // LDS int4
#endif

// ---------------------------------------------------------------------------
// PCEN: per (b,h) row, EMA recurrence over the contiguous W axis (64 steps).
// ---------------------------------------------------------------------------
__global__ void pcen_kernel(const float* __restrict__ x,
                            const float* __restrict__ log_s,
                            const float* __restrict__ log_alpha,
                            const float* __restrict__ log_delta,
                            const float* __restrict__ log_r,
                            _Float16* __restrict__ out)
{
    int t = blockIdx.x * blockDim.x + threadIdx.x;   // t = b*64 + h
    if (t >= 2048 * 64) return;
    const float s     = expf(log_s[0]);
    const float alpha = expf(log_alpha[0]);
    const float delta = expf(log_delta[0]);
    const float r     = expf(log_r[0]);
    const float dr    = powf(delta, r);
    const float* row  = x + (size_t)t * 64;
    _Float16* orow    = out + (size_t)t * 64;
    float m = 0.f;
    for (int w = 0; w < 64; ++w) {
        float xv = row[w];
        m = (1.f - s) * m + s * xv;
        float v = powf(xv / powf(m + 1e-6f, alpha) + delta, r) - dr;
        orow[w] = (_Float16)v;
    }
}

// ---------------------------------------------------------------------------
// Fuse conv bias + BN (eval) into per-channel scale/bias.
// sb layout: [0..15]=scale1 [16..31]=bias1 [32..63]=scale2 [64..95]=bias2
// ---------------------------------------------------------------------------
__global__ void prep_scalebias(const float* c1b, const float* g1, const float* b1,
                               const float* m1, const float* v1,
                               const float* c2b, const float* g2, const float* b2,
                               const float* m2, const float* v2,
                               float* __restrict__ sb)
{
    int t = threadIdx.x;
    if (t < 16) {
        float sc = 0.f, bi = 0.f;
        if (t < 15) {
            float inv = g1[t] * rsqrtf(v1[t] + 1e-5f);
            sc = inv; bi = c1b[t] * inv + b1[t] - m1[t] * inv;
        }
        sb[t] = sc; sb[16 + t] = bi;
    } else if (t < 48) {
        int c = t - 16;
        float sc = 0.f, bi = 0.f;
        if (c < 30) {
            float inv = g2[c] * rsqrtf(v2[c] + 1e-5f);
            sc = inv; bi = c2b[c] * inv + b2[c] - m2[c] * inv;
        }
        sb[32 + c] = sc; sb[64 + c] = bi;
    }
}

// ---------------------------------------------------------------------------
// Pre-pack B matrices (K x 16 tiles, K padded to multiples of 32) into the
// per-lane WMMA f16 B-fragment layout:
//   slot j of lane L holds B[K, N] with K = kb*32 + 8*(L>=16) + j + (j&8), N = L&15.
// frag index == global thread index (layout ((ntile*nkb+kb)*32+lane)*16+j).
// w is row-major [Nvalid, Kvalid]; B[k][n] = w[n*Kvalid + k].
// ---------------------------------------------------------------------------
__global__ void prep_bfrag(const float* __restrict__ w, _Float16* __restrict__ frag,
                           int Kvalid, int Nvalid, int nkb, long total)
{
    long t = (long)blockIdx.x * blockDim.x + threadIdx.x;
    if (t >= total) return;
    int  j     = (int)(t & 15);
    int  lane  = (int)((t >> 4) & 31);
    long rest  = t >> 9;
    int  kb    = (int)(rest % nkb);
    int  ntile = (int)(rest / nkb);
    int  K = kb * 32 + ((lane >> 4) << 3) + j + (j & 8);
    int  N = ntile * 16 + (lane & 15);
    float v = (K < Kvalid && N < Nvalid) ? w[(size_t)N * Kvalid + K] : 0.f;
    frag[t] = (_Float16)v;
}

// ---------------------------------------------------------------------------
// Implicit-GEMM 7x7 VALID conv via WMMA with LDS-staged input.
// One block = one image: stage Cin*Hin*Win f16 into LDS (async copy engine if
// available, else B128 loads+stores), then 8 waves sweep all
// (pixel-tile x channel-tile) WMMA tiles, gathering A-fragments from LDS.
// Epilogue fuses BN+ReLU.
// A layout (16-bit A 16x32): lanes<16 hold K {0..7,16..23},
// lanes>=16 hold {8..15,24..31}; M = lane&15.
// ---------------------------------------------------------------------------
__global__ void __launch_bounds__(256) conv_wmma_lds(
    const _Float16* __restrict__ in, const _Float16* __restrict__ bfrag,
    const float* __restrict__ sb_scale, const float* __restrict__ sb_bias,
    _Float16* __restrict__ out,
    int Cin, int Hin, int Win, int Hout, int Wout, int Cout,
    int nkb, int Kvalid, int ntilesM, int ntilesN, long in_img_stride)
{
    __shared__ __align__(16) _Float16 smem[16384];   // 32 KB (max image 25 KB)
    const int tid  = threadIdx.x;
    const int lane = tid & 31;
    const int wid  = tid >> 5;
    const int b    = blockIdx.x;
    const int npix = Hout * Wout;
    const int T    = Cin * Hin * Win;

    // ---- cooperative stage: global -> LDS ----
    const _Float16* src = in + (size_t)b * in_img_stride;
    int nv = T >> 3;                                  // 16-byte chunks
#if HAVE_ASYNC_LDS
    {
        gv4i* g = (gv4i*)src;                         // addrspacecast to global
        lv4i* l = (lv4i*)smem;                        // addrspacecast to LDS
        for (int i = tid; i < nv; i += 256)
            __builtin_amdgcn_global_load_async_to_lds_b128(g + i, l + i, 0, 0);
        __builtin_amdgcn_s_wait_asynccnt(0);
    }
#else
    {
        const uint4* s4 = (const uint4*)src;
        uint4*       d4 = (uint4*)smem;
        for (int i = tid; i < nv; i += 256) d4[i] = s4[i];
    }
#endif
    for (int i = (nv << 3) + tid; i < T; i += 256) smem[i] = src[i];
    __syncthreads();

    const int lhalf  = lane >> 4;
    const int HW     = Hin * Win;
    const int nTiles = ntilesM * ntilesN;
    for (int wt = wid; wt < nTiles; wt += 8) {
        int tile  = wt % ntilesM;
        int ntile = wt / ntilesM;
        int pix = tile * 16 + (lane & 15);
        int p   = pix < npix ? pix : 0;
        int oy  = p / Wout, ox = p - oy * Wout;
        const _Float16* bf = bfrag + (size_t)ntile * nkb * 512;

        v8f acc = {};
        for (int kb = 0; kb < nkb; ++kb) {
            v16h a;
            int kbase = kb * 32 + lhalf * 8;
            #pragma unroll
            for (int j = 0; j < 16; ++j) {
                int K = kbase + j + (j & 8);
                _Float16 v = (_Float16)0.f;
                if (K < Kvalid) {
                    int cch = K / 49;
                    int rem = K - cch * 49;
                    int ky  = rem / 7, kx = rem - ky * 7;
                    v = smem[cch * HW + (oy + ky) * Win + (ox + kx)];
                }
                a[j] = v;
            }
            v16h bm = *(const v16h*)(bf + ((size_t)(kb * 32 + lane) << 4));
            acc = __builtin_amdgcn_wmma_f32_16x16x32_f16(false, a, false, bm,
                                                         (short)0, acc, false, false);
        }

        const int   N     = lane & 15;
        const int   ch    = ntile * 16 + N;
        const float scale = sb_scale[ch];
        const float bias  = sb_bias[ch];
        const int   Mbase = lhalf * 8;
        #pragma unroll
        for (int rr = 0; rr < 8; ++rr) {
            int pr = tile * 16 + Mbase + rr;
            if (ch < Cout && pr < npix) {
                float v = acc[rr] * scale + bias;
                out[((size_t)b * Cout + ch) * npix + pr] = (_Float16)(v > 0.f ? v : 0.f);
            }
        }
    }
}

// ---------------------------------------------------------------------------
// 2x2 stride-2 max pool (VALID), f16 -> f16, padded output image stride.
// ---------------------------------------------------------------------------
__global__ void maxpool2_kernel(const _Float16* __restrict__ in,
                                _Float16* __restrict__ out,
                                int C, int Hin, int Win, int Hout, int Wout,
                                long out_img_stride, long total)
{
    long t = (long)blockIdx.x * blockDim.x + threadIdx.x;
    if (t >= total) return;
    int ow = (int)(t % Wout); long r = t / Wout;
    int oh = (int)(r % Hout); r /= Hout;
    int c  = (int)(r % C);    long b = r / C;
    const _Float16* s = in + (size_t)(b * C + c) * Hin * Win;
    int y = 2 * oh, x = 2 * ow;
    float a0 = (float)s[(size_t)y * Win + x];
    float a1 = (float)s[(size_t)y * Win + x + 1];
    float a2 = (float)s[(size_t)(y + 1) * Win + x];
    float a3 = (float)s[(size_t)(y + 1) * Win + x + 1];
    out[(size_t)b * out_img_stride + ((size_t)c * Hout + oh) * Wout + ow] =
        (_Float16)fmaxf(fmaxf(a0, a1), fmaxf(a2, a3));
}

// ---------------------------------------------------------------------------
// FC1 GEMM: out[2048,200] = relu(act[2048,3630] * W^T + b). M=batch tiles,
// N=13 tiles of 16, K=3630 -> 114 WMMAs per wave. A rows contiguous f16
// (stride padded to 3632 halves so fragment runs are 16B-aligned).
// ---------------------------------------------------------------------------
__global__ void fc1_wmma(const _Float16* __restrict__ act,
                         const _Float16* __restrict__ bfrag,
                         const float* __restrict__ fc1_b,
                         float* __restrict__ out)
{
    const int lane  = threadIdx.x & 31;
    int ntile = blockIdx.x * (blockDim.x >> 5) + (threadIdx.x >> 5);
    if (ntile >= 13) return;                     // wave-uniform exit
    const int mtile = blockIdx.y;                // 0..127
    const int lhalf = lane >> 4;
    const int brow  = mtile * 16 + (lane & 15);

    const _Float16* arow = act + (size_t)brow * P2STRIDE;
    const _Float16* bf   = bfrag + (size_t)ntile * 114 * 512;

    v8f acc = {};
    for (int kb = 0; kb < 114; ++kb) {
        v16h a;
        int kbase = kb * 32 + lhalf * 8;
        __builtin_prefetch(arow + kbase + 128, 0, 1);   // -> global_prefetch_b8
        if (kb < 113) {                          // fully in-range: contiguous runs
            #pragma unroll
            for (int j = 0; j < 8; ++j)  a[j] = arow[kbase + j];
            #pragma unroll
            for (int j = 8; j < 16; ++j) a[j] = arow[kbase + 8 + j];
        } else {                                 // K tail 3616..3647, valid < 3630
            #pragma unroll
            for (int j = 0; j < 16; ++j) {
                int K = kbase + j + (j & 8);
                a[j] = (K < 3630) ? arow[K] : (_Float16)0.f;
            }
        }
        v16h bm = *(const v16h*)(bf + ((size_t)(kb * 32 + lane) << 4));
        acc = __builtin_amdgcn_wmma_f32_16x16x32_f16(false, a, false, bm,
                                                     (short)0, acc, false, false);
    }

    const int   N    = lane & 15;
    const int   n    = ntile * 16 + N;
    const float bias = (n < 200) ? fc1_b[n] : 0.f;
    const int   Mbase = lhalf * 8;
    #pragma unroll
    for (int rr = 0; rr < 8; ++rr) {
        int br = mtile * 16 + Mbase + rr;
        if (n < 200) {
            float v = acc[rr] + bias;
            out[(size_t)br * 200 + n] = v > 0.f ? v : 0.f;
        }
    }
}

// ---------------------------------------------------------------------------
// FC2 (200 -> 2) + sigmoid. Tiny: one thread per output element.
// ---------------------------------------------------------------------------
__global__ void fc2_kernel(const float* __restrict__ fc1out,
                           const float* __restrict__ w,
                           const float* __restrict__ bias,
                           float* __restrict__ out)
{
    int t = blockIdx.x * blockDim.x + threadIdx.x;
    if (t >= 2048 * 2) return;
    int o = t & 1, b = t >> 1;
    const float* a  = fc1out + (size_t)b * 200;
    const float* wr = w + (size_t)o * 200;
    float s = bias[o];
    for (int k = 0; k < 200; ++k) s += a[k] * wr[k];
    out[t] = 1.f / (1.f + expf(-s));
}

// ---------------------------------------------------------------------------
extern "C" void kernel_launch(void* const* d_in, const int* in_sizes, int n_in,
                              void* d_out, int out_size, void* d_ws, size_t ws_size,
                              hipStream_t stream)
{
    (void)in_sizes; (void)n_in; (void)out_size; (void)ws_size;

    const float* x         = (const float*)d_in[0];
    const float* log_s     = (const float*)d_in[1];
    const float* log_alpha = (const float*)d_in[2];
    const float* log_delta = (const float*)d_in[3];
    const float* log_r     = (const float*)d_in[4];
    const float* conv1_w   = (const float*)d_in[5];
    const float* conv1_b   = (const float*)d_in[6];
    const float* bn1_g     = (const float*)d_in[7];
    const float* bn1_b     = (const float*)d_in[8];
    const float* bn1_m     = (const float*)d_in[9];
    const float* bn1_v     = (const float*)d_in[10];
    const float* conv2_w   = (const float*)d_in[11];
    const float* conv2_b   = (const float*)d_in[12];
    const float* bn2_g     = (const float*)d_in[13];
    const float* bn2_b     = (const float*)d_in[14];
    const float* bn2_m     = (const float*)d_in[15];
    const float* bn2_v     = (const float*)d_in[16];
    const float* fc1_w     = (const float*)d_in[17];
    const float* fc1_b     = (const float*)d_in[18];
    const float* fc2_w     = (const float*)d_in[19];
    const float* fc2_b     = (const float*)d_in[20];

    char* ws = (char*)d_ws;
    size_t off = 0;
    auto carve = [&](size_t bytes) -> char* {
        off = (off + 255) & ~(size_t)255;
        char* p = ws + off; off += bytes; return p;
    };
    _Float16* pcen16 = (_Float16*)carve(2048ull * 4096 * 2);        // PCEN f16
    _Float16* c1out  = (_Float16*)carve(2048ull * 15 * 3364 * 2);   // conv1 act
    _Float16* p1out  = (_Float16*)carve(2048ull * P1STRIDE * 2);    // pool1 (padded)
    _Float16* c2out  = (_Float16*)carve(2048ull * 30 * 529 * 2);    // conv2 act
    _Float16* p2out  = (_Float16*)carve(2048ull * P2STRIDE * 2);    // pool2 (padded)
    float*    fc1out = (float*)   carve(2048ull * 200 * 4);
    _Float16* bf1    = (_Float16*)carve(1ull  * 2   * 512 * 2);     // conv1 B frags
    _Float16* bf2    = (_Float16*)carve(2ull  * 23  * 512 * 2);     // conv2 B frags
    _Float16* bffc1  = (_Float16*)carve(13ull * 114 * 512 * 2);     // fc1 B frags
    float*    sb     = (float*)   carve(96 * 4);                    // fused BN

    // --- weight / scale prep ---
    prep_scalebias<<<1, 64, 0, stream>>>(conv1_b, bn1_g, bn1_b, bn1_m, bn1_v,
                                         conv2_b, bn2_g, bn2_b, bn2_m, bn2_v, sb);
    { long tot = 1L * 2 * 512;
      prep_bfrag<<<(int)((tot + 255) / 256), 256, 0, stream>>>(conv1_w, bf1, 49, 15, 2, tot); }
    { long tot = 2L * 23 * 512;
      prep_bfrag<<<(int)((tot + 255) / 256), 256, 0, stream>>>(conv2_w, bf2, 735, 30, 23, tot); }
    { long tot = 13L * 114 * 512;
      prep_bfrag<<<(int)((tot + 255) / 256), 256, 0, stream>>>(fc1_w, bffc1, 3630, 200, 114, tot); }

    // --- PCEN ---
    pcen_kernel<<<(2048 * 64 + 255) / 256, 256, 0, stream>>>(
        x, log_s, log_alpha, log_delta, log_r, pcen16);

    // --- conv1: LDS-staged image (8 KB), 211 pixel tiles, 2 WMMAs/tile ---
    conv_wmma_lds<<<2048, 256, 0, stream>>>(
        pcen16, bf1, sb + 0, sb + 16, c1out,
        1, 64, 64, 58, 58, 15, 2, 49, 211, 1, 4096);
    maxpool2_kernel<<<(int)((2048L * 15 * 841 + 255) / 256), 256, 0, stream>>>(
        c1out, p1out, 15, 58, 58, 29, 29, (long)P1STRIDE, 2048L * 15 * 841);

    // --- conv2: LDS-staged image (25 KB), 34 pix-tiles x 2 ch-tiles, 23 WMMAs ---
    conv_wmma_lds<<<2048, 256, 0, stream>>>(
        p1out, bf2, sb + 32, sb + 64, c2out,
        15, 29, 29, 23, 23, 30, 23, 735, 34, 2, (long)P1STRIDE);
    maxpool2_kernel<<<(int)((2048L * 30 * 121 + 255) / 256), 256, 0, stream>>>(
        c2out, p2out, 30, 23, 23, 11, 11, (long)P2STRIDE, 2048L * 30 * 121);

    // --- fc1 GEMM (114 WMMAs/wave) + fc2 + sigmoid ---
    fc1_wmma<<<dim3(2, 128), 256, 0, stream>>>(p2out, bffc1, fc1_b, fc1out);
    fc2_kernel<<<(4096 + 255) / 256, 256, 0, stream>>>(fc1out, fc2_w, fc2_b, (float*)d_out);
}
